// Transformer_Prediction_13846974562412
// MI455X (gfx1250) — matrix operand order repaired and checked by reading, not compiled, not run on hardware
//
#include <hip/hip_runtime.h>
#include <math.h>

// ---------------------------------------------------------------------------
// y = LN2( gelu( LN1(x) @ W^T + bW ) )    x:[16384,2048] f32, W:[2048,2048]
//
//  K1: cast W -> bf16 (8 MB, L2-resident)
//  K2: LN1 + bf16 cast -> Hn (64 MB ws)
//  K3: GEMM, M=32 rows/WG, N=2048, 512 thr (16 waves -> 4 waves/SIMD so the
//      128 acc VGPRs/wave fit without scratch spills):
//        TDM double-buffered k-slabs (A 32x32, B 2048x32 bf16) into LDS,
//        per wave 2 M-tiles x 8 N-tiles (each B frag feeds 2 wmmas),
//        batched frag loads -> back-to-back v_wmma_f32_16x16x32_bf16,
//        bias+gelu in-register, LN2 via shuffle + ds_add_f32, direct store.
// ---------------------------------------------------------------------------

#define D          2048
#define KC         32
#define MT         32                      // rows per workgroup
#define NT         512                     // threads per workgroup (16 waves)
#define NSTEPS     (D / KC)                // 64
#define BSTR_BYTES 72                      // 32 bf16 (64B) + 8B TDM pad (conflict-free, 8B aligned)
#define ABUF_BYTES (MT * BSTR_BYTES)       // 2304
#define BBUF_BYTES (D  * BSTR_BYTES)       // 147456
#define SMEM_GEMM  (2 * BBUF_BYTES + 2 * ABUF_BYTES)   // 299520 B < 320 KB

typedef __bf16 bf16_t;
typedef __attribute__((ext_vector_type(16))) __bf16       v16bf;
typedef __attribute__((ext_vector_type(8)))  float        v8f;
typedef __attribute__((ext_vector_type(4)))  unsigned int u32x4;
typedef __attribute__((ext_vector_type(8)))  int          i32x8;
typedef __attribute__((ext_vector_type(4)))  int          i32x4;

union Frag { uint2 u[4]; v16bf v; };       // 32 B = 16 bf16

#if defined(__AMDGCN__) && __has_builtin(__builtin_amdgcn_tensor_load_to_lds)
#define HAVE_TDM 1
#else
#define HAVE_TDM 0
#endif

__device__ __forceinline__ bf16_t f2bf(float f) {
    union { float f; unsigned u; } in; in.f = f;
    unsigned r = in.u + 0x7FFFu + ((in.u >> 16) & 1u);            // RNE
    union { unsigned short s; bf16_t b; } o; o.s = (unsigned short)(r >> 16);
    return o.b;
}

#if HAVE_TDM
// 2-D TDM load: tile (tile0 x tile1) of a row-major tensor (stride0 elems/row,
// 2-byte elems) -> LDS at lds_addr, padded +2 dwords per 16 dwords (row stride 72B).
__device__ __forceinline__ void tdm_load_2d(unsigned lds_addr, const void* gptr,
                                            unsigned td0, unsigned td1,
                                            unsigned tile0, unsigned tile1,
                                            unsigned stride0) {
    unsigned long long ga = (unsigned long long)gptr;
    // D# group0: count=1 | lds_addr | global_addr[56:0] | type=2
    u32x4 g0 = { 0x1u,
                 lds_addr,
                 (unsigned)ga,
                 (unsigned)(ga >> 32) | (2u << 30) };
    // D# group1: data_size=2B, pad_enable, pad_interval=16dw(code3), pad_amount=2dw(code1)
    i32x8 g1 = { (int)((1u << 16) | (1u << 20) | (3u << 22) | (1u << 25)),
                 (int)((td0 & 0xFFFFu) << 16),                     // tensor_dim0[15:0]
                 (int)((td0 >> 16) | ((td1 & 0xFFFFu) << 16)),     // dim0[31:16] | dim1[15:0]
                 (int)((td1 >> 16) | (tile0 << 16)),               // dim1[31:16] | tile_dim0
                 (int)(tile1 & 0xFFFFu),                           // tile_dim1 (tile_dim2=0)
                 (int)stride0,                                     // tensor_dim0_stride[31:0]
                 0, 0 };
    i32x4 z4 = { 0, 0, 0, 0 };
#if defined(__clang_major__) && (__clang_major__ >= 23)
    i32x8 z8 = { 0, 0, 0, 0, 0, 0, 0, 0 };
    __builtin_amdgcn_tensor_load_to_lds(g0, g1, z4, z4, z8, 0);
#else
    __builtin_amdgcn_tensor_load_to_lds(g0, g1, z4, z4, 0);
#endif
}
#endif

// Stage the k-slab pair (A: MT x KC from Hn, B: D x KC from Wb) into buffer `sel`.
__device__ __forceinline__ void stage_slabs(char* smem, const bf16_t* __restrict__ Hn,
                                            const bf16_t* __restrict__ Wb,
                                            int row0, int k0, int sel, int tid, int wave) {
#if HAVE_TDM
    if (wave == 0) {
        unsigned ldsB = (unsigned)(unsigned long long)(void*)(smem + sel * BBUF_BYTES);
        unsigned ldsA = (unsigned)(unsigned long long)(void*)(smem + 2 * BBUF_BYTES + sel * ABUF_BYTES);
        tdm_load_2d(ldsB, Wb + k0,                    D, D,     KC, D,  D);
        tdm_load_2d(ldsA, Hn + (size_t)row0 * D + k0, D, 16384, KC, MT, D);
    }
#else
    char* bB = smem + sel * BBUF_BYTES;
    char* bA = smem + 2 * BBUF_BYTES + sel * ABUF_BYTES;
    for (int r = tid; r < D; r += NT) {
        const uint2* src = (const uint2*)(Wb + (size_t)r * D + k0);
        uint2* dst = (uint2*)(bB + r * BSTR_BYTES);
        #pragma unroll
        for (int j = 0; j < 8; ++j) dst[j] = src[j];
    }
    if (tid < MT) {
        const uint2* src = (const uint2*)(Hn + (size_t)(row0 + tid) * D + k0);
        uint2* dst = (uint2*)(bA + tid * BSTR_BYTES);
        #pragma unroll
        for (int j = 0; j < 8; ++j) dst[j] = src[j];
    }
#endif
}

__device__ __forceinline__ void stage_wait(int wave) {
#if HAVE_TDM
    if (wave == 0) __builtin_amdgcn_s_wait_tensorcnt(0);
#endif
}

// --- K1: W -> bf16 ----------------------------------------------------------
__global__ void __launch_bounds__(256) cast_w_bf16(const float* __restrict__ W,
                                                   bf16_t* __restrict__ Wb) {
    int i = (blockIdx.x * 256 + threadIdx.x) * 4;
    float4 w = *(const float4*)(W + i);
    union { bf16_t h[4]; uint2 u; } pk;
    pk.h[0] = f2bf(w.x); pk.h[1] = f2bf(w.y);
    pk.h[2] = f2bf(w.z); pk.h[3] = f2bf(w.w);
    *(uint2*)(Wb + i) = pk.u;
}

// --- K2: LN1 + bf16 cast (one wave per row) ---------------------------------
__global__ void __launch_bounds__(256)
ln1_cast(const float* __restrict__ x, const float* __restrict__ g1,
         const float* __restrict__ b1, bf16_t* __restrict__ Hn) {
    const int lane = threadIdx.x & 31, wave = threadIdx.x >> 5;
    const int row  = blockIdx.x * 8 + wave;
    const float* xr = x + (size_t)row * D;
    float4 vb[16];
    float s = 0.f, ss = 0.f;
    #pragma unroll
    for (int j = 0; j < 16; ++j) {
        vb[j] = *(const float4*)(xr + 4 * (lane + 32 * j));
        s  += vb[j].x + vb[j].y + vb[j].z + vb[j].w;
        ss += vb[j].x * vb[j].x + vb[j].y * vb[j].y + vb[j].z * vb[j].z + vb[j].w * vb[j].w;
    }
    #pragma unroll
    for (int off = 16; off >= 1; off >>= 1) {
        s += __shfl_xor(s, off); ss += __shfl_xor(ss, off);
    }
    float mu = s * (1.0f / D);
    float rs = rsqrtf(ss * (1.0f / D) - mu * mu + 1e-5f);
    bf16_t* hr = Hn + (size_t)row * D;
    #pragma unroll
    for (int j = 0; j < 16; ++j) {
        int c = 4 * (lane + 32 * j);
        float4 gg = *(const float4*)(g1 + c);
        float4 bb = *(const float4*)(b1 + c);
        union { bf16_t h[4]; uint2 u; } pk;
        pk.h[0] = f2bf((vb[j].x - mu) * rs * gg.x + bb.x);
        pk.h[1] = f2bf((vb[j].y - mu) * rs * gg.y + bb.y);
        pk.h[2] = f2bf((vb[j].z - mu) * rs * gg.z + bb.z);
        pk.h[3] = f2bf((vb[j].w - mu) * rs * gg.w + bb.w);
        *(uint2*)(hr + c) = pk.u;
    }
}

// --- K3: WMMA GEMM + bias + gelu + fused LN2 --------------------------------
__global__ void __launch_bounds__(NT)
Transformer_Prediction_13846974562412_kernel(const bf16_t* __restrict__ Hn,
                                             const bf16_t* __restrict__ Wb,
                                             const float*  __restrict__ bW,
                                             const float*  __restrict__ g2,
                                             const float*  __restrict__ b2,
                                             float* __restrict__ out) {
    extern __shared__ char smem[];
    __shared__ float s_sum[MT], s_ssum[MT], s_mu[MT], s_rs[MT];

    const int tid   = threadIdx.x;
    const int lane  = tid & 31;
    const int wave  = tid >> 5;            // 0..15
    const int row0  = blockIdx.x * MT;
    const int nb    = wave * 128;          // this wave's N window: 8 tiles of 16
    const int nlane = lane & 15;
    const int half  = lane >> 4;
    const int kbAb  = half * 16;           // A frag K byte base (8 bf16)
    const int kbBb  = half * 16;           // B frag K byte base (8 bf16)

    if (tid < MT) { s_sum[tid] = 0.f; s_ssum[tid] = 0.f; }

    stage_slabs(smem, Hn, Wb, row0, 0, 0, tid, wave);
    stage_wait(wave);
    __syncthreads();

    v8f acc[16];   // [mt*8 + j]: mt in {0,1} (rows mt*16..), j = N-tile 0..7
    #pragma unroll
    for (int t = 0; t < 16; ++t)
        acc[t] = (v8f){0.f, 0.f, 0.f, 0.f, 0.f, 0.f, 0.f, 0.f};

    for (int i = 0; i < NSTEPS; ++i) {
        const int cur = i & 1;
        if (i + 1 < NSTEPS)
            stage_slabs(smem, Hn, Wb, row0, (i + 1) * KC, 1 - cur, tid, wave);

        const char* aB = smem + 2 * BBUF_BYTES + cur * ABUF_BYTES;
        const char* bB = smem + cur * BBUF_BYTES;

        // Both A frags up front (reused by all 8 N-tiles)
        Frag a0, a1;
        {
            const char* ap0 = aB + nlane * BSTR_BYTES + kbAb;
            a0.u[0] = *(const uint2*)(ap0);      a0.u[1] = *(const uint2*)(ap0 + 8);
            a0.u[2] = *(const uint2*)(ap0 + 32); a0.u[3] = *(const uint2*)(ap0 + 40);
            const char* ap1 = ap0 + 16 * BSTR_BYTES;
            a1.u[0] = *(const uint2*)(ap1);      a1.u[1] = *(const uint2*)(ap1 + 8);
            a1.u[2] = *(const uint2*)(ap1 + 32); a1.u[3] = *(const uint2*)(ap1 + 40);
        }

        // B frags in chunks of 4; each feeds two wmmas (mt=0,1) back-to-back
        #pragma unroll
        for (int c = 0; c < 2; ++c) {
            Frag b[4];
            #pragma unroll
            for (int j = 0; j < 4; ++j) {
                const char* bp = bB + (nb + (c * 4 + j) * 16 + nlane) * BSTR_BYTES + kbBb;
                b[j].u[0] = *(const uint2*)(bp);
                b[j].u[1] = *(const uint2*)(bp + 8);
                b[j].u[2] = *(const uint2*)(bp + 32);
                b[j].u[3] = *(const uint2*)(bp + 40);
            }
            #pragma unroll
            for (int j = 0; j < 4; ++j) {
                acc[c * 4 + j] = __builtin_amdgcn_wmma_f32_16x16x32_bf16(
                    false, a0.v, false, b[j].v, (short)0, acc[c * 4 + j], false, false);
                acc[8 + c * 4 + j] = __builtin_amdgcn_wmma_f32_16x16x32_bf16(
                    false, a1.v, false, b[j].v, (short)0, acc[8 + c * 4 + j], false, false);
            }
        }

        if (i + 1 < NSTEPS) stage_wait(wave);
        __syncthreads();
    }

    // ---- bias + gelu in-register, per-row partial sums ----
    float s8[16], ss8[16];                 // [mt*8 + v]
    #pragma unroll
    for (int v = 0; v < 16; ++v) { s8[v] = 0.f; ss8[v] = 0.f; }

    #pragma unroll
    for (int j = 0; j < 8; ++j) {
        float bw = bW[nb + j * 16 + nlane];
        #pragma unroll
        for (int mt = 0; mt < 2; ++mt) {
            #pragma unroll
            for (int v = 0; v < 8; ++v) {
                float val = acc[mt * 8 + j][v] + bw;
                val = 0.5f * val * (1.0f + erff(val * 0.70710678118654752f));
                acc[mt * 8 + j][v] = val;
                s8[mt * 8 + v] += val; ss8[mt * 8 + v] += val * val;
            }
        }
    }
    // reduce across the 16 lanes of each N-group (row fixed per half-wave)
    #pragma unroll
    for (int v = 0; v < 16; ++v) {
        #pragma unroll
        for (int off = 8; off >= 1; off >>= 1) {
            s8[v]  += __shfl_xor(s8[v], off);
            ss8[v] += __shfl_xor(ss8[v], off);
        }
    }
    if (nlane == 0) {
        #pragma unroll
        for (int mt = 0; mt < 2; ++mt) {
            #pragma unroll
            for (int v = 0; v < 8; ++v) {
                int m = mt * 16 + half * 8 + v;
                atomicAdd(&s_sum[m],  s8[mt * 8 + v]);    // ds_add_f32 cross-wave
                atomicAdd(&s_ssum[m], ss8[mt * 8 + v]);
            }
        }
    }
    __syncthreads();

    if (tid < MT) {
        float mu  = s_sum[tid] * (1.0f / D);
        float var = s_ssum[tid] * (1.0f / D) - mu * mu;
        s_mu[tid] = mu;
        s_rs[tid] = rsqrtf(var + 1e-5f);
    }
    __syncthreads();

    // ---- LN2 normalize + store ----
    #pragma unroll
    for (int j = 0; j < 8; ++j) {
        int n = nb + j * 16 + nlane;
        float gg = g2[n], bb = b2[n];
        #pragma unroll
        for (int mt = 0; mt < 2; ++mt) {
            #pragma unroll
            for (int v = 0; v < 8; ++v) {
                int m = mt * 16 + half * 8 + v;
                out[(size_t)(row0 + m) * D + n] =
                    (acc[mt * 8 + j][v] - s_mu[m]) * s_rs[m] * gg + bb;
            }
        }
    }
}

extern "C" void kernel_launch(void* const* d_in, const int* in_sizes, int n_in,
                              void* d_out, int out_size, void* d_ws, size_t ws_size,
                              hipStream_t stream) {
    const float* x  = (const float*)d_in[0];
    const float* W  = (const float*)d_in[1];
    const float* bW = (const float*)d_in[2];
    const float* g1 = (const float*)d_in[3];
    const float* b1 = (const float*)d_in[4];
    const float* g2 = (const float*)d_in[5];
    const float* b2 = (const float*)d_in[6];
    float* out = (float*)d_out;

    bf16_t* Wb = (bf16_t*)d_ws;                                   // 8 MB
    bf16_t* Hn = (bf16_t*)((char*)d_ws + (size_t)D * D * 2);      // 64 MB

    cast_w_bf16<<<(D * D) / (256 * 4), 256, 0, stream>>>(W, Wb);
    ln1_cast<<<16384 / 8, 256, 0, stream>>>(x, g1, b1, Hn);
    Transformer_Prediction_13846974562412_kernel<<<16384 / MT, NT, SMEM_GEMM, stream>>>(
        Hn, Wb, bW, g2, b2, out);
}
